// BaseMLP_61160334295155
// MI455X (gfx1250) — matrix operand order repaired and checked by reading, not compiled, not run on hardware
//
#include <hip/hip_runtime.h>

typedef __attribute__((ext_vector_type(16))) _Float16 v16h;
typedef __attribute__((ext_vector_type(8)))  float    v8f;
typedef __attribute__((ext_vector_type(4)))  float    f4;

constexpr int kD  = 256;   // nodes
constexpr int kH  = 64;    // hidden
constexpr int kP  = 2;     // params out
constexpr int kBS = 1024;  // batch
constexpr int kTT = 16;    // t per block (one t per wave, 16 waves)
constexpr int kNB = 32;    // batch per block (2 n-tiles of 16)
constexpr int kKC = 32;    // K chunk (WMMA K)
constexpr int kNC = kD / kKC;  // 8 K-chunks
constexpr int kThreads = 512;

// ---- fallback: build a 16x32 f16 A-fragment from f32 weights in-flight ----
__device__ inline v16h load_a16_f32(const float* __restrict__ rp) {
    f4 a0 = *(const f4*)(rp);
    f4 a1 = *(const f4*)(rp + 4);
    f4 a2 = *(const f4*)(rp + 8);
    f4 a3 = *(const f4*)(rp + 12);
    v16h af;
#pragma unroll
    for (int e = 0; e < 4; ++e) {
        af[e]      = (_Float16)a0[e];
        af[4 + e]  = (_Float16)a1[e];
        af[8 + e]  = (_Float16)a2[e];
        af[12 + e] = (_Float16)a3[e];
    }
    return af;
}

// ---- pre-pack kernels: f32 weights -> f16, laid out in A-fragment order ----
// W0p slot id g = ((t*8 + kc)*4 + it)*32 + lane ; each slot = 16 halves (32B).
__global__ __launch_bounds__(256)
void pack_w0(const float* __restrict__ W0, _Float16* __restrict__ W0p) {
    const int g    = blockIdx.x * 256 + threadIdx.x;   // 0 .. 256K-1
    const int lane = g & 31;
    const int it   = (g >> 5) & 3;
    const int kc   = (g >> 7) & 7;
    const int t    = g >> 10;
    const int ln   = lane & 15;
    const int hi   = lane >> 4;
    const float* row = W0 + ((size_t)t * kH + it * 16 + ln) * kD + kc * kKC + hi * 16;
    _Float16* dst = W0p + (size_t)g * 16;
#pragma unroll
    for (int e = 0; e < 8; ++e) dst[e]     = (_Float16)row[e];
#pragma unroll
    for (int e = 0; e < 8; ++e) dst[8 + e] = (_Float16)row[8 + e];
}

// W1p slot id g = ((t*2 + kk)*4 + it)*32 + lane
__global__ __launch_bounds__(256)
void pack_w1(const float* __restrict__ W1, _Float16* __restrict__ W1p) {
    const int g    = blockIdx.x * 256 + threadIdx.x;   // 0 .. 64K-1
    const int lane = g & 31;
    const int it   = (g >> 5) & 3;
    const int kk   = (g >> 7) & 1;
    const int t    = g >> 8;
    const int ln   = lane & 15;
    const int hi   = lane >> 4;
    const float* row = W1 + ((size_t)t * kH + it * 16 + ln) * kH + kk * 32 + hi * 16;
    _Float16* dst = W1p + (size_t)g * 16;
#pragma unroll
    for (int e = 0; e < 8; ++e) dst[e]     = (_Float16)row[e];
#pragma unroll
    for (int e = 0; e < 8; ++e) dst[8 + e] = (_Float16)row[8 + e];
}

template <bool PACKED>
__global__ __launch_bounds__(kThreads, 1)
void fused_node_mlp(const float* __restrict__ x,  const float* __restrict__ M,
                    const float* __restrict__ W0, const float* __restrict__ b0,
                    const float* __restrict__ W1, const float* __restrict__ b1,
                    const float* __restrict__ W2, const float* __restrict__ b2,
                    const _Float16* __restrict__ W0p, const _Float16* __restrict__ W1p,
                    float* __restrict__ out)
{
    __shared__ __align__(32) _Float16 xmL[2][kTT * kNB * kKC]; // [buf][t][b][j] 2x32 KB
    __shared__ __align__(32) _Float16 hL [kTT * kNB * kH ];    // [t][b][i]      64 KB
    __shared__ float b0s[kTT * kH];                            // 4 KB
    __shared__ float b1s[kTT * kH];                            // 4 KB

    const int tid  = threadIdx.x;
    const int lane = tid & 31;
    const int wave = tid >> 5;           // 0..15  (= t_local)
    const int ln   = lane & 15;          // n index within tile / row within A
    const int hi16 = lane >> 4;          // half-wave select (k split)
    const int t0   = blockIdx.x * kTT;   // node-tile base
    const int b0g  = blockIdx.y * kNB;   // batch-tile base
    const int tl   = wave;
    const int t    = t0 + tl;

    for (int idx = tid; idx < kTT * kH; idx += kThreads) {
        b0s[idx] = b0[t0 * kH + idx];
        b1s[idx] = b1[t0 * kH + idx];
    }

    // -------- staging decode (fixed per thread): t-quad sQ, batch bS, j-octet jh ----
    const int sQ = tid & 3;            // t = t0 + sQ*4 + e
    const int bS = (tid >> 2) & 31;    // batch row 0..31
    const int jh = tid >> 7;           // j octet (jl = jh*8 + q)
    const float* mBase = M + (size_t)(b0g + bS) * kD * kD + (size_t)(jh * 8) * kD
                           + t0 + sQ * 4;
    const float* xBase = x + (size_t)(b0g + bS) * kD + jh * 8;
    const size_t sOff  = (size_t)(sQ * 4) * kNB * kKC + bS * kKC + jh * 8;

    v8f acc[2][4] = {};  // [n-tile][i-tile]
    f4    mv[8];         // staged M values (next chunk)
    float xv[8];         // staged x values (next chunk)

    auto stage_load = [&](int kc) {
        const float* mP = mBase + (size_t)kc * kKC * kD;
        const float* xP = xBase + kc * kKC;
#pragma unroll
        for (int q = 0; q < 8; ++q) {
            mv[q] = __builtin_nontemporal_load((const f4*)(mP + (size_t)q * kD));
            xv[q] = xP[q];
        }
    };
    auto stage_store = [&](int kc, int buf) {
        const int j0 = kc * kKC + jh * 8;
        _Float16* sB = &xmL[buf][0] + sOff;
#pragma unroll
        for (int e = 0; e < 4; ++e) {              // e outer: 8 contiguous halves/store
            const int tg = t0 + sQ * 4 + e;
#pragma unroll
            for (int q = 0; q < 8; ++q) {
                float v = (j0 + q == tg) ? 0.0f : mv[q][e] * xv[q];  // adj = 1 - eye
                sB[(size_t)e * kNB * kKC + q] = (_Float16)v;
            }
        }
    };
    auto compute0 = [&](int kc, int buf) {
        v16h bf0 = *(const v16h*)&xmL[buf][(tl * kNB + ln)      * kKC + hi16 * 16];
        v16h bf1 = *(const v16h*)&xmL[buf][(tl * kNB + 16 + ln) * kKC + hi16 * 16];
#pragma unroll
        for (int it = 0; it < 4; ++it) {
            v16h af;
            if (PACKED) {
                af = *(const v16h*)(W0p + ((size_t)((t * 8 + kc) * 4 + it) * 32 + lane) * 16);
            } else {
                af = load_a16_f32(W0 + (size_t)t * kH * kD
                                     + (size_t)(it * 16 + ln) * kD + kc * kKC + hi16 * 16);
            }
            acc[0][it] = __builtin_amdgcn_wmma_f32_16x16x32_f16(
                false, af, false, bf0, (short)0, acc[0][it], false, false);
            acc[1][it] = __builtin_amdgcn_wmma_f32_16x16x32_f16(
                false, af, false, bf1, (short)0, acc[1][it], false, false);
        }
    };

    // ---------------- Layer 0: software-pipelined over 8 K-chunks ----------------
    stage_load(0);
    stage_store(0, 0);
    __syncthreads();
    for (int kc = 0; kc < kNC - 1; ++kc) {
        stage_load(kc + 1);               // NT loads in flight during WMMAs
        compute0(kc, kc & 1);
        stage_store(kc + 1, (kc + 1) & 1);
        __syncthreads();                  // one barrier per chunk
    }
    compute0(kNC - 1, (kNC - 1) & 1);

    // Epilogue 0: + b0, leaky_relu, f16 -> LDS [t][b][i]
#pragma unroll
    for (int nt = 0; nt < 2; ++nt) {
        const int b = nt * 16 + ln;
#pragma unroll
        for (int it = 0; it < 4; ++it) {
#pragma unroll
            for (int r = 0; r < 8; ++r) {
                const int i = it * 16 + r + hi16 * 8;   // C layout: M = r + 8*(lane>=16)
                float v = acc[nt][it][r] + b0s[tl * kH + i];
                v = v > 0.0f ? v : 0.01f * v;
                hL[(tl * kNB + b) * kH + i] = (_Float16)v;
            }
        }
    }

    // ---------------- Layer 1: h1 = W1[t] (64x64) x h0 (64 x 32) ----------------
    v8f acc1[2][4] = {};
#pragma unroll
    for (int kk = 0; kk < 2; ++kk) {
        v16h bf0 = *(const v16h*)&hL[(tl * kNB + ln)      * kH + kk * 32 + hi16 * 16];
        v16h bf1 = *(const v16h*)&hL[(tl * kNB + 16 + ln) * kH + kk * 32 + hi16 * 16];
#pragma unroll
        for (int it = 0; it < 4; ++it) {
            v16h af;
            if (PACKED) {
                af = *(const v16h*)(W1p + ((size_t)((t * 2 + kk) * 4 + it) * 32 + lane) * 16);
            } else {
                af = load_a16_f32(W1 + (size_t)t * kH * kH
                                     + (size_t)(it * 16 + ln) * kH + kk * 32 + hi16 * 16);
            }
            acc1[0][it] = __builtin_amdgcn_wmma_f32_16x16x32_f16(
                false, af, false, bf0, (short)0, acc1[0][it], false, false);
            acc1[1][it] = __builtin_amdgcn_wmma_f32_16x16x32_f16(
                false, af, false, bf1, (short)0, acc1[1][it], false, false);
        }
    }

    // Epilogue 1: + b1, leaky_relu -> overwrite hL (wave-private region, DS in-order)
#pragma unroll
    for (int nt = 0; nt < 2; ++nt) {
        const int b = nt * 16 + ln;
#pragma unroll
        for (int it = 0; it < 4; ++it) {
#pragma unroll
            for (int r = 0; r < 8; ++r) {
                const int i = it * 16 + r + hi16 * 8;
                float v = acc1[nt][it][r] + b1s[tl * kH + i];
                v = v > 0.0f ? v : 0.01f * v;
                hL[(tl * kNB + b) * kH + i] = (_Float16)v;
            }
        }
    }

    // ---------------- Layer 2: out = W2[t] (2x64) x h1 (64 x 32) + b2 ----------------
    const _Float16* hr = &hL[(tl * kNB + lane) * kH];
#pragma unroll
    for (int p = 0; p < kP; ++p) {
        const float* w2r = W2 + (size_t)t * kP * kH + p * kH;
        float sum = 0.0f;
#pragma unroll
        for (int k = 0; k < kH; k += 4) {
            f4 w = *(const f4*)(w2r + k);
            sum += w[0] * (float)hr[k]     + w[1] * (float)hr[k + 1]
                 + w[2] * (float)hr[k + 2] + w[3] * (float)hr[k + 3];
        }
        sum += b2[t * kP + p];
        __builtin_nontemporal_store(sum,
            &out[(size_t)(b0g + lane) * (kD * kP) + t * kP + p]);
    }
}

extern "C" void kernel_launch(void* const* d_in, const int* in_sizes, int n_in,
                              void* d_out, int out_size, void* d_ws, size_t ws_size,
                              hipStream_t stream) {
    const float* x  = (const float*)d_in[0];
    const float* M  = (const float*)d_in[1];
    // d_in[2] = adj (ones - eye) folded into the (j != t) predicate
    const float* W0 = (const float*)d_in[3];
    const float* b0 = (const float*)d_in[4];
    const float* W1 = (const float*)d_in[5];
    const float* b1 = (const float*)d_in[6];
    const float* W2 = (const float*)d_in[7];
    const float* b2 = (const float*)d_in[8];
    float* out = (float*)d_out;

    const size_t w0Elems = (size_t)kD * kH * kD;   // 4M
    const size_t w1Elems = (size_t)kD * kH * kH;   // 1M
    const size_t need    = (w0Elems + w1Elems) * sizeof(_Float16);  // 10 MB

    dim3 grid(kD / kTT, kBS / kNB);   // 16 x 32 blocks, 512 threads each

    if (d_ws != nullptr && ws_size >= need) {
        _Float16* W0p = (_Float16*)d_ws;
        _Float16* W1p = W0p + w0Elems;
        pack_w0<<<(int)(w0Elems / 16 / 256), 256, 0, stream>>>(W0, W0p);
        pack_w1<<<(int)(w1Elems / 16 / 256), 256, 0, stream>>>(W1, W1p);
        fused_node_mlp<true><<<grid, kThreads, 0, stream>>>(
            x, M, W0, b0, W1, b1, W2, b2, W0p, W1p, out);
    } else {
        fused_node_mlp<false><<<grid, kThreads, 0, stream>>>(
            x, M, W0, b0, W1, b1, W2, b2, nullptr, nullptr, out);
    }
}
